// ScaledDotProductAttention_61649960567146
// MI455X (gfx1250) — matrix operand order repaired and checked by reading, not compiled, not run on hardware
//
#include <hip/hip_runtime.h>
#include <hip/hip_fp16.h>
#include <math.h>

typedef __attribute__((ext_vector_type(16))) _Float16 v16h;
typedef __attribute__((ext_vector_type(8)))  float    v8f;

#define BB 2
#define HH 16
#define SS 2048
#define DD 64

union AFrag { v16h h; uint32_t u[8]; };

__global__ __launch_bounds__(128)
void fa_fwd_kernel(const float* __restrict__ Q,
                   const float* __restrict__ K,
                   const float* __restrict__ V,
                   const int*   __restrict__ Mk,
                   float*       __restrict__ Out)
{
    // V^T tile: [d][key], stride 34 halves (conflict-free u32 gathers)
    __shared__ _Float16 VtS[DD * 34];
    // per-wave P tile: [m][k], stride 34 halves
    __shared__ _Float16 PlS[4][16 * 34];

    const int tid  = threadIdx.x;
    const int wave = tid >> 5;
    const int lane = tid & 31;
    const int g    = lane >> 4;     // half-wave group (WMMA layout selector)
    const int n    = lane & 15;     // column index within 16
    const int m    = n;             // A-fragment row owned by this lane

    const int bh    = blockIdx.y;         // 0..B*H-1
    const int b     = bh / HH;
    const int qbase = blockIdx.x * 64 + wave * 16;

    const float* Qp = Q  + ((size_t)bh * SS + qbase) * DD;
    const float* Kp = K  + (size_t)bh * SS * DD;
    const float* Vp = V  + (size_t)bh * SS * DD;
    const int*   Mp = Mk + (size_t)b * SS * SS;

    // ---- load Q rows once, build A fragments (pre-scaled by 1/sqrt(D)=0.125) ----
    float qrow[DD];
    #pragma unroll
    for (int i = 0; i < DD; i += 4) {
        float4 t = *(const float4*)(Qp + (size_t)m * DD + i);
        qrow[i] = t.x; qrow[i+1] = t.y; qrow[i+2] = t.z; qrow[i+3] = t.w;
    }
    v16h a0, a1;   // A fragments: d = 0..31 and d = 32..63
    #pragma unroll
    for (int vv = 0; vv < 8; ++vv) {
        const int kb = 2*vv;
        #pragma unroll
        for (int p = 0; p < 2; ++p) {
            const float lo  = qrow[kb + p]      * 0.125f;
            const float hi  = qrow[kb + p + 16] * 0.125f;
            const float lo2 = qrow[kb + p + 32] * 0.125f;
            const float hi2 = qrow[kb + p + 48] * 0.125f;
            a0[2*vv + p] = (_Float16)(g ? hi  : lo);
            a1[2*vv + p] = (_Float16)(g ? hi2 : lo2);
        }
    }

    v8f o0 = {}, o1 = {}, o2 = {}, o3 = {};   // O accumulator, 16 x 64 (4 N-tiles)
    float rmax[8], rsum[8];
    #pragma unroll
    for (int r = 0; r < 8; ++r) { rmax[r] = -INFINITY; rsum[r] = 0.f; }

    _Float16* Pl   = PlS[wave];
    uint32_t* Pl32 = reinterpret_cast<uint32_t*>(Pl);
    uint32_t* Vt32 = reinterpret_cast<uint32_t*>(VtS);

    for (int j0 = 0; j0 < SS; j0 += 32) {
        __syncthreads();   // protect Vt from overwrite while other waves read
        // ---- stage V tile (32 keys x 64 dims) into LDS transposed, f16 ----
        {
            const int key = tid >> 2;
            const int seg = tid & 3;
            const float* vp = Vp + ((size_t)(j0 + key)) * DD + seg * 16;
            #pragma unroll
            for (int i = 0; i < 16; i += 4) {
                float4 t = *(const float4*)(vp + i);
                const int d = seg * 16 + i;
                VtS[(d+0)*34 + key] = (_Float16)t.x;
                VtS[(d+1)*34 + key] = (_Float16)t.y;
                VtS[(d+2)*34 + key] = (_Float16)t.z;
                VtS[(d+3)*34 + key] = (_Float16)t.w;
            }
        }
        if (j0 + 32 < SS) {   // uniform branch: EXEC unchanged
            __builtin_prefetch(Vp + (size_t)(j0 + 32 + lane) * DD, 0, 1);
            __builtin_prefetch(Kp + (size_t)(j0 + 32 + lane) * DD, 0, 1);
        }
        __syncthreads();

        // ---- scores: c0 = keys [j0, j0+16), c1 = keys [j0+16, j0+32) ----
        v8f c0 = {}, c1 = {};
        #pragma unroll
        for (int c = 0; c < 2; ++c) {
            // B fragment of K^T: lane holds 16 consecutive d of one K row
            const float* kb0 = Kp + (size_t)(j0 + 16*c + n) * DD + 16*g;
            AFrag bf0, bf1;
            #pragma unroll
            for (int i = 0; i < 16; i += 4) {
                float4 t0 = *(const float4*)(kb0 + i);
                float4 t1 = *(const float4*)(kb0 + 32 + i);
                bf0.h[i]   = (_Float16)t0.x; bf0.h[i+1] = (_Float16)t0.y;
                bf0.h[i+2] = (_Float16)t0.z; bf0.h[i+3] = (_Float16)t0.w;
                bf1.h[i]   = (_Float16)t1.x; bf1.h[i+1] = (_Float16)t1.y;
                bf1.h[i+2] = (_Float16)t1.z; bf1.h[i+3] = (_Float16)t1.w;
            }
            v8f acc = (c == 0) ? c0 : c1;
            acc = __builtin_amdgcn_wmma_f32_16x16x32_f16(false, a0, false, bf0.h,
                                                         (short)0, acc, false, false);
            acc = __builtin_amdgcn_wmma_f32_16x16x32_f16(false, a1, false, bf1.h,
                                                         (short)0, acc, false, false);
            if (c == 0) c0 = acc; else c1 = acc;
        }

        // ---- mask + online softmax (row r+8g lives across this half-wave) ----
        #pragma unroll
        for (int r = 0; r < 8; ++r) {
            const size_t mrow = (size_t)(qbase + r + 8*g) * SS + j0 + n;
            c0[r] = (Mp[mrow]      != 0) ? c0[r] : -1e9f;
            c1[r] = (Mp[mrow + 16] != 0) ? c1[r] : -1e9f;
            float t = fmaxf(c0[r], c1[r]);
            t = fmaxf(t, __shfl_xor(t, 1, 32));
            t = fmaxf(t, __shfl_xor(t, 2, 32));
            t = fmaxf(t, __shfl_xor(t, 4, 32));
            t = fmaxf(t, __shfl_xor(t, 8, 32));
            const float nm    = fmaxf(rmax[r], t);
            const float alpha = __expf(rmax[r] - nm);
            rmax[r] = nm;
            const float p0 = __expf(c0[r] - nm);
            const float p1 = __expf(c1[r] - nm);
            float ps = p0 + p1;
            ps += __shfl_xor(ps, 1, 32);
            ps += __shfl_xor(ps, 2, 32);
            ps += __shfl_xor(ps, 4, 32);
            ps += __shfl_xor(ps, 8, 32);
            rsum[r] = rsum[r] * alpha + ps;
            o0[r] *= alpha; o1[r] *= alpha; o2[r] *= alpha; o3[r] *= alpha;
            Pl[(r + 8*g)*34 + n]      = (_Float16)p0;   // C-layout -> [m][k] in LDS
            Pl[(r + 8*g)*34 + 16 + n] = (_Float16)p1;
        }

        // same-wave LDS is in-order; wait for stores before fragment gathers
        asm volatile("s_wait_dscnt 0" ::: "memory");

        // ---- gather P as A-fragment (row m, consecutive-k pairs as u32) ----
        AFrag pa;
        #pragma unroll
        for (int vv = 0; vv < 8; ++vv) {
            const int idx = m*17 + vv + 8*g;
            pa.u[vv] = Pl32[idx];
        }

        // ---- O += P * V over 4 output d-tiles ----
        #pragma unroll
        for (int t = 0; t < 4; ++t) {
            AFrag vb;   // B fragment: column d=16t+n, key pairs 2v+16g
            #pragma unroll
            for (int vv = 0; vv < 8; ++vv)
                vb.u[vv] = Vt32[(16*t + n)*17 + vv + 8*g];
            v8f acc = (t==0) ? o0 : (t==1) ? o1 : (t==2) ? o2 : o3;
            acc = __builtin_amdgcn_wmma_f32_16x16x32_f16(false, pa.h, false, vb.h,
                                                         (short)0, acc, false, false);
            if (t==0) o0=acc; else if (t==1) o1=acc; else if (t==2) o2=acc; else o3=acc;
        }
    }

    // ---- epilogue: normalize and store fp32, coalesced per half-wave ----
    float* Op = Out + ((size_t)bh * SS + qbase) * DD;
    #pragma unroll
    for (int r = 0; r < 8; ++r) {
        const float inv = 1.0f / rsum[r];
        const int row = r + 8*g;
        Op[(size_t)row*DD +  0 + n] = o0[r] * inv;
        Op[(size_t)row*DD + 16 + n] = o1[r] * inv;
        Op[(size_t)row*DD + 32 + n] = o2[r] * inv;
        Op[(size_t)row*DD + 48 + n] = o3[r] * inv;
    }
}

extern "C" void kernel_launch(void* const* d_in, const int* in_sizes, int n_in,
                              void* d_out, int out_size, void* d_ws, size_t ws_size,
                              hipStream_t stream) {
    const float* Q  = (const float*)d_in[0];
    const float* K  = (const float*)d_in[1];
    const float* V  = (const float*)d_in[2];
    const int*   Mk = (const int*)d_in[3];
    float* Out = (float*)d_out;
    (void)in_sizes; (void)n_in; (void)out_size; (void)d_ws; (void)ws_size;

    dim3 grid(SS / 64, BB * HH);   // 32 x 32 blocks, 4 waves each
    fa_fwd_kernel<<<grid, dim3(128), 0, stream>>>(Q, K, V, Mk, Out);
}
